// AttentionBackward_25941602468158
// MI455X (gfx1250) — matrix operand order repaired and checked by reading, not compiled, not run on hardware
//
#include <hip/hip_runtime.h>

// Problem constants (compile-time, from reference)
#define MM 2048
#define NN 1024
#define DD 64
#define HH 16

typedef __attribute__((ext_vector_type(4)))  _Float16 h4;
typedef __attribute__((ext_vector_type(8)))  _Float16 h8;
typedef __attribute__((ext_vector_type(16))) _Float16 h16;
typedef __attribute__((ext_vector_type(8)))  float    v8f;

// Load a 16-half fragment as two 16-byte chunks (two *_load_b128).
__device__ __forceinline__ h16 frag2(const _Float16* p0, const _Float16* p1) {
  h16 r;
  ((h8*)&r)[0] = *(const h8*)p0;
  ((h8*)&r)[1] = *(const h8*)p1;
  return r;
}

__device__ __forceinline__ v8f wmma16(h16 a, h16 b, v8f c) {
  return __builtin_amdgcn_wmma_f32_16x16x32_f16(
      /*neg_a=*/false, a, /*neg_b=*/false, b,
      /*c_mod=*/(short)0, c, /*reuse_a=*/false, /*reuse_b=*/false);
}

// ---------------------------------------------------------------------------
// Kernel 1: X (f32, M x N) -> Xh (f16), vectorized x4
// ---------------------------------------------------------------------------
__global__ __launch_bounds__(256) void k_convert_x(const float* __restrict__ X,
                                                   _Float16* __restrict__ Xh) {
  int i = (blockIdx.x * 256 + threadIdx.x) * 4;
  float4 v = *(const float4*)(X + i);
  h4 o;
  o.x = (_Float16)v.x; o.y = (_Float16)v.y;
  o.z = (_Float16)v.z; o.w = (_Float16)v.w;
  *(h4*)(Xh + i) = o;
}

// ---------------------------------------------------------------------------
// Kernel 2: W (f32, K x N) -> Wt (f16, N x K), LDS-tiled transpose.
// blockIdx.z selects which of WQ/WK/WV.
// ---------------------------------------------------------------------------
__global__ __launch_bounds__(256) void k_transpose_w(
    const float* __restrict__ WQ, const float* __restrict__ WK,
    const float* __restrict__ WV, _Float16* __restrict__ WQt,
    _Float16* __restrict__ WKt, _Float16* __restrict__ WVt) {
  __shared__ float tile[32][33];
  const float* W = (blockIdx.z == 0) ? WQ : (blockIdx.z == 1) ? WK : WV;
  _Float16* Wt   = (blockIdx.z == 0) ? WQt : (blockIdx.z == 1) ? WKt : WVt;

  int x = blockIdx.x * 32 + threadIdx.x;
  int y = blockIdx.y * 32 + threadIdx.y;
#pragma unroll
  for (int i = 0; i < 32; i += 8)
    tile[threadIdx.y + i][threadIdx.x] = W[(size_t)(y + i) * NN + x];
  __syncthreads();
  int xo = blockIdx.y * 32 + threadIdx.x;
  int yo = blockIdx.x * 32 + threadIdx.y;
#pragma unroll
  for (int i = 0; i < 32; i += 8)
    Wt[(size_t)(yo + i) * NN + xo] = (_Float16)tile[threadIdx.x][threadIdx.y + i];
}

// ---------------------------------------------------------------------------
// Kernel 3: projections. C = Xh @ W (via Wt). WHICH is compile-time so each
// instantiation has a branch-free epilogue.
// 8 waves per WG; each wave owns a 32x32 output tile (2x2 WMMA tiles).
//   WHICH=0: write Q1 (f32, to d_out) and Qh (f16)
//   WHICH=1: write Kh (f16, row-major)
//   WHICH=2: write Vt (f16, per-head d-major: Vt[(h*64+d)*M + m])
// ---------------------------------------------------------------------------
template <int WHICH>
__global__ __launch_bounds__(256) void k_proj(
    const _Float16* __restrict__ Xh, const _Float16* __restrict__ Wt,
    float* __restrict__ Cf32, _Float16* __restrict__ Cf16) {
  const int wave = threadIdx.x >> 5;
  const int lane = threadIdx.x & 31;
  const int lr = lane & 15;        // row/col within 16
  const int lh = lane >> 4;        // half-wave select
  const int mrow = blockIdx.y * 64 + (wave >> 2) * 32;
  const int ncol = blockIdx.x * 128 + (wave & 3) * 32;

  v8f acc[2][2] = {};
  for (int k = 0; k < NN; k += 32) {
    h16 a[2], b[2];
#pragma unroll
    for (int rt = 0; rt < 2; ++rt) {
      const _Float16* pa = Xh + (size_t)(mrow + rt * 16 + lr) * NN + k + 8 * lh;
      a[rt] = frag2(pa, pa + 16);
    }
#pragma unroll
    for (int ct = 0; ct < 2; ++ct) {
      const _Float16* pb = Wt + (size_t)(ncol + ct * 16 + lr) * NN + k + 8 * lh;
      b[ct] = frag2(pb, pb + 16);
    }
#pragma unroll
    for (int rt = 0; rt < 2; ++rt)
#pragma unroll
      for (int ct = 0; ct < 2; ++ct)
        acc[rt][ct] = wmma16(a[rt], b[ct], acc[rt][ct]);
  }

  // Branch-free epilogue per WHICH.
#pragma unroll
  for (int rt = 0; rt < 2; ++rt) {
    const int mbase = mrow + rt * 16 + 8 * lh;  // row for j=0
#pragma unroll
    for (int ct = 0; ct < 2; ++ct) {
      const int n = ncol + ct * 16 + lr;
      if (WHICH == 2) {
        // V: transposed write Vt[n*MM + m]
        _Float16* pv = Cf16 + (size_t)n * MM + mbase;
#pragma unroll
        for (int j = 0; j < 8; ++j) pv[j] = (_Float16)acc[rt][ct][j];
      } else {
        const size_t base = (size_t)mbase * NN + n;
#pragma unroll
        for (int j = 0; j < 8; ++j) {
          float v = acc[rt][ct][j];
          Cf16[base + (size_t)j * NN] = (_Float16)v;
          if (WHICH == 0) Cf32[base + (size_t)j * NN] = v;
        }
      }
    }
  }
}

// ---------------------------------------------------------------------------
// Kernel 4: streaming attention. One WG = one head x 64 query rows (4 waves,
// 16 rows each). Flash-style loop over key blocks of 32; scores never spilled.
// ---------------------------------------------------------------------------
__global__ __launch_bounds__(128) void k_attn(
    const _Float16* __restrict__ Qh, const _Float16* __restrict__ Kh,
    const _Float16* __restrict__ Vt, float* __restrict__ O2) {
  __shared__ _Float16 lds_p[4][16 * 32];  // per-wave P relayout scratch

  const int wave = threadIdx.x >> 5;
  const int lane = threadIdx.x & 31;
  const int lr = lane & 15;
  const int lh = lane >> 4;
  const int h = blockIdx.y;
  const int hoff = h * DD;
  const int q0 = blockIdx.x * 64 + wave * 16;

  // Q A-fragments (d 0..31 and d 32..63), resident for whole kernel
  const _Float16* pq = Qh + (size_t)(q0 + lr) * NN + hoff + 8 * lh;
  h16 aQ0 = frag2(pq, pq + 16);
  h16 aQ1 = frag2(pq + 32, pq + 48);

  v8f o[4] = {};
  v8f rs = {};
  _Float16* P = lds_p[wave];

  for (int kb = 0; kb < MM; kb += 32) {
    // S = Q K^T for two 16-key tiles, chained over D=64
    v8f s[2];
#pragma unroll
    for (int t = 0; t < 2; ++t) {
      const _Float16* pk = Kh + (size_t)(kb + t * 16 + lr) * NN + hoff + 8 * lh;
      h16 b0 = frag2(pk, pk + 16);
      h16 b1 = frag2(pk + 32, pk + 48);
      v8f z = {};
      z = wmma16(aQ0, b0, z);
      s[t] = wmma16(aQ1, b1, z);
    }
    // exp + rowsum + stash f16 P tile into LDS for C->A relayout
#pragma unroll
    for (int t = 0; t < 2; ++t)
#pragma unroll
      for (int j = 0; j < 8; ++j) {
        float e = __expf(s[t][j]);
        rs[j] += e;
        P[(j + 8 * lh) * 32 + t * 16 + lr] = (_Float16)e;
      }
    // reload P as A-fragment (rows in-lane)
    const _Float16* pp = P + lr * 32 + 8 * lh;
    h16 aP = frag2(pp, pp + 16);

    // O += P @ V over the 4 d-tiles (Vt is d-major => contiguous B loads)
#pragma unroll
    for (int td = 0; td < 4; ++td) {
      const _Float16* pv =
          Vt + (size_t)(hoff + td * 16 + lr) * MM + kb + 8 * lh;
      h16 bV = frag2(pv, pv + 16);
      o[td] = wmma16(aP, bV, o[td]);
    }
  }

  // softmax denominator: reduce rowsum across the 16 lanes of each half-wave,
  // then take one reciprocal per row.
  v8f inv;
#pragma unroll
  for (int j = 0; j < 8; ++j) {
    float r = rs[j];
#pragma unroll
    for (int msk = 8; msk >= 1; msk >>= 1) r += __shfl_xor(r, msk, 16);
    inv[j] = 1.0f / r;
  }

  // normalize + store O2 (lanes 0..15 of each half cover 16 consecutive cols)
  float* po = O2 + (size_t)(q0 + 8 * lh) * NN + hoff + lr;
#pragma unroll
  for (int td = 0; td < 4; ++td)
#pragma unroll
    for (int j = 0; j < 8; ++j)
      po[(size_t)j * NN + td * 16] = o[td][j] * inv[j];
}

// ---------------------------------------------------------------------------
extern "C" void kernel_launch(void* const* d_in, const int* in_sizes, int n_in,
                              void* d_out, int out_size, void* d_ws,
                              size_t ws_size, hipStream_t stream) {
  (void)in_sizes; (void)n_in; (void)out_size; (void)ws_size;
  const float* X  = (const float*)d_in[0];
  const float* WQ = (const float*)d_in[1];
  const float* WK = (const float*)d_in[2];
  const float* WV = (const float*)d_in[3];

  float* O2 = (float*)d_out;                         // (M, N) f32
  float* Q1 = (float*)d_out + (size_t)MM * NN;       // (M, N) f32

  // Workspace carve-up (22 MB total)
  char* ws = (char*)d_ws;
  _Float16* Xh  = (_Float16*)ws; ws += (size_t)MM * NN * 2;  // 4 MB
  _Float16* WQt = (_Float16*)ws; ws += (size_t)NN * NN * 2;  // 2 MB
  _Float16* WKt = (_Float16*)ws; ws += (size_t)NN * NN * 2;  // 2 MB
  _Float16* WVt = (_Float16*)ws; ws += (size_t)NN * NN * 2;  // 2 MB
  _Float16* Qhp = (_Float16*)ws; ws += (size_t)MM * NN * 2;  // 4 MB
  _Float16* Khp = (_Float16*)ws; ws += (size_t)MM * NN * 2;  // 4 MB
  _Float16* Vtp = (_Float16*)ws; ws += (size_t)MM * NN * 2;  // 4 MB

  k_convert_x<<<(MM * NN) / (256 * 4), 256, 0, stream>>>(X, Xh);
  k_transpose_w<<<dim3(NN / 32, NN / 32, 3), dim3(32, 8), 0, stream>>>(
      WQ, WK, WV, WQt, WKt, WVt);

  const dim3 pg(NN / 128, MM / 64);
  k_proj<0><<<pg, 256, 0, stream>>>(Xh, WQt, Q1, Qhp);
  k_proj<1><<<pg, 256, 0, stream>>>(Xh, WKt, nullptr, Khp);
  k_proj<2><<<pg, 256, 0, stream>>>(Xh, WVt, nullptr, Vtp);

  k_attn<<<dim3(MM / 64, HH), 128, 0, stream>>>(Qhp, Khp, Vtp, O2);
}
